// siiformer_NLattmil_48077863911757
// MI455X (gfx1250) — matrix operand order repaired and checked by reading, not compiled, not run on hardware
//
#include <hip/hip_runtime.h>
#include <hip/hip_bf16.h>
#include <stdint.h>

#define N_TOK   16384
#define DIM     1024
#define HD      512
#define HD2     256
#define TOPK_K  100
#define M_SEL   4915   // int(0.3 * 16384)

#define SIM_ROWS 64            // rows per block in similarity kernel (4 row-tiles)
#define SIM_LDS_BYTES (SIM_ROWS*HD*2 /*qs*/ + SIM_ROWS*256*4 /*chunk*/ + SIM_ROWS*TOPK_K*4 /*top*/)

typedef __attribute__((ext_vector_type(16))) _Float16 v16h;
typedef __attribute__((ext_vector_type(8)))  _Float16 v8h;
typedef __attribute__((ext_vector_type(8)))  float    v8f;

__device__ __forceinline__ v8f wmma_f16(v16h a, v16h b, v8f c) {
    return __builtin_amdgcn_wmma_f32_16x16x32_f16(
        false, a, false, b, (short)0, c, false, false);
}

// A fragment (16x32 f16) from an f32 row pointer (per-lane row), convert on the fly.
// Layout (ISA 7.12.2): lane<16: e0-7 -> k0..7, e8-15 -> k16..23 ; lane>=16: +8.
__device__ __forceinline__ v16h load_a_f32(const float* __restrict__ rowptr, int k0, int lane) {
    int off = (lane < 16) ? 0 : 8;
    const float* p0 = rowptr + k0 + off;
    const float* p1 = rowptr + k0 + 16 + off;
    v16h a;
#pragma unroll
    for (int e = 0; e < 8; ++e) a[e] = (_Float16)p0[e];
#pragma unroll
    for (int e = 0; e < 8; ++e) a[8 + e] = (_Float16)p1[e];
    return a;
}

// A fragment from f16 row pointer (global or LDS): two contiguous v8h groups.
__device__ __forceinline__ v16h load_a_f16(const _Float16* rowptr, int k0, int lane) {
    int off = (lane < 16) ? 0 : 8;
    union { v16h v; v8h h[2]; } u;
    u.h[0] = *(const v8h*)(rowptr + k0 + off);
    u.h[1] = *(const v8h*)(rowptr + k0 + 16 + off);
    return u.v;
}

// B fragment (32x16 f16): per-lane column n, e0-7 -> k0+off..+7, e8-15 -> k0+16+off..+7, off = (lane<16?0:8).
// colrow must point at the [n][K] row for this lane's n.
__device__ __forceinline__ v16h load_b_f16(const _Float16* __restrict__ colrow, int k0, int lane) {
    int off = (lane < 16) ? 0 : 8;
    union { v16h v; v8h h[2]; } u;
    u.h[0] = *(const v8h*)(colrow + k0 + off);
    u.h[1] = *(const v8h*)(colrow + k0 + 16 + off);
    return u.v;
}

// ---------------- Kernel 0: W (DIM x HD) f32 -> WT (HD x DIM) f16 ----------------
__global__ __launch_bounds__(256) void k_wt(const float* __restrict__ W, _Float16* __restrict__ WT) {
    int idx = blockIdx.x * 256 + threadIdx.x;      // over DIM*HD = 524288
    int k = idx >> 9;                              // / HD
    int n = idx & (HD - 1);
    WT[n * DIM + k] = (_Float16)W[idx];
}

// ---------------- Kernel 1: outh = f16( l2norm( tanh(x @ W + b) ) ) ----------------
// grid.x = N/16 row tiles, block = 256 (8 waves, each wave 4 column tiles of 16)
__global__ __launch_bounds__(256) void k_proj(const float* __restrict__ x,
                                              const _Float16* __restrict__ WT,
                                              const float* __restrict__ bias,
                                              _Float16* __restrict__ outh) {
    __shared__ float tile[16 * HD];
    __shared__ float ssq[16];
    int tid = threadIdx.x, wave = tid >> 5, lane = tid & 31;
    int row0 = blockIdx.x * 16;
    int m = lane & 15, n = lane & 15;
    const float* xrow = x + (size_t)(row0 + m) * DIM;

    v8f acc[4];
#pragma unroll
    for (int t = 0; t < 4; ++t) acc[t] = (v8f){};
    for (int k0 = 0; k0 < DIM; k0 += 32) {
        v16h a = load_a_f32(xrow, k0, lane);
#pragma unroll
        for (int t = 0; t < 4; ++t) {
            int n0 = (wave * 4 + t) * 16;
            v16h b = load_b_f16(WT + (size_t)(n0 + n) * DIM, k0, lane);
            acc[t] = wmma_f16(a, b, acc[t]);
        }
    }
#pragma unroll
    for (int t = 0; t < 4; ++t) {
        int n0 = (wave * 4 + t) * 16;
        float bb = bias[n0 + n];
#pragma unroll
        for (int r = 0; r < 8; ++r) {
            int mm = r + (lane < 16 ? 0 : 8);
            tile[mm * HD + n0 + n] = tanhf(acc[t][r] + bb);
        }
    }
    __syncthreads();
    if (tid < 16) ssq[tid] = 0.0f;
    __syncthreads();
    {   // 16 threads per row, 32 cols each
        int rr = tid >> 4, seg = tid & 15;
        float s = 0.0f;
        for (int j = 0; j < 32; ++j) { float v = tile[rr * HD + seg * 32 + j]; s += v * v; }
        atomicAdd(&ssq[rr], s);
    }
    __syncthreads();
    for (int i = tid; i < 16 * HD; i += 256) {
        int rr = i >> 9;
        float inv = 1.0f / fmaxf(sqrtf(ssq[rr]), 1e-12f);
        outh[(size_t)(row0 + rr) * HD + (i & (HD - 1))] = (_Float16)(tile[i] * inv);
    }
}

// ---------------- Kernel 2: similarity GEMM + fused streaming top-100 mean ----------------
// block = 512 (16 waves) handles 64 rows (4 row-tiles); iterates 64 chunks of 256 columns.
// Each wave owns one 16-col tile per chunk and computes 4 row-tiles against it, so every
// B fragment fetched from L2 feeds 4 WMMAs. Dynamic LDS (156 KB) within 320 KB/WGP.
__global__ __launch_bounds__(512) void k_sim_topk(const _Float16* __restrict__ qh,
                                                  const _Float16* __restrict__ kh,
                                                  float* __restrict__ A1) {
    extern __shared__ char smem[];
    _Float16* qs   = (_Float16*)smem;                              // 64*512 f16 = 64 KB
    float*    chunk= (float*)(smem + SIM_ROWS * HD * 2);           // 64*256 f32 = 64 KB
    float*    top  = (float*)(smem + SIM_ROWS * HD * 2 + SIM_ROWS * 256 * 4); // 25.6 KB

    int tid = threadIdx.x, wave = tid >> 5, lane = tid & 31;
    int row0 = blockIdx.x * SIM_ROWS;

    {   // stage 64-row q tile into LDS (contiguous copy)
        const uint32_t* src = (const uint32_t*)(qh + (size_t)row0 * HD);
        uint32_t* dst = (uint32_t*)qs;
        for (int i = tid; i < SIM_ROWS * HD / 2; i += 512) dst[i] = src[i];
    }
    for (int i = tid; i < SIM_ROWS * TOPK_K; i += 512) top[i] = -3.0e38f;
    __syncthreads();

    int m = lane & 15, n = lane & 15;
    float vmin[4];
#pragma unroll
    for (int r = 0; r < 4; ++r) vmin[r] = -3.0e38f;

    for (int s = 0; s < N_TOK / 256; ++s) {
        int col0 = s * 256 + wave * 16;
        // warm L2->WGP path for next chunk's B rows (global_prefetch_b8)
        if (s + 1 < N_TOK / 256)
            __builtin_prefetch(kh + (size_t)(col0 + 256 + n) * HD, 0, 1);

        v8f acc[4];
#pragma unroll
        for (int rt = 0; rt < 4; ++rt) acc[rt] = (v8f){};
        for (int k0 = 0; k0 < HD; k0 += 32) {
            v16h b = load_b_f16(kh + (size_t)(col0 + n) * HD, k0, lane);
#pragma unroll
            for (int rt = 0; rt < 4; ++rt) {
                v16h a = load_a_f16(qs + (rt * 16 + m) * HD, k0, lane);
                acc[rt] = wmma_f16(a, b, acc[rt]);
            }
        }
#pragma unroll
        for (int rt = 0; rt < 4; ++rt) {
#pragma unroll
            for (int r = 0; r < 8; ++r) {
                int mm = rt * 16 + r + (lane < 16 ? 0 : 8);
                chunk[mm * 256 + wave * 16 + n] = acc[rt][r];
            }
        }
        __syncthreads();
        // merge: wave w handles rows {w, w+16, w+32, w+48} (ballot-filtered, wave-uniform flow)
#pragma unroll
        for (int rr = 0; rr < 4; ++rr) {
            int row = wave + rr * 16;
            float* rowTop = top + row * TOPK_K;
            float vm = vmin[rr];
            for (int i = lane; i < 256; i += 32) {
                float v = chunk[row * 256 + i];
                unsigned long long ball = __ballot(v > vm);
                while (ball) {
                    int srcl = __ffsll((unsigned long long)ball) - 1;
                    ball &= ball - 1;
                    float cand = __shfl(v, srcl, 32);
                    if (cand > vm) {            // recheck: vm may have risen
                        // cooperative argmin over rowTop[0..99]
                        float lv = 3.0e38f; int li = 0;
                        for (int j = lane; j < TOPK_K; j += 32) {
                            float t = rowTop[j];
                            if (t < lv) { lv = t; li = j; }
                        }
#pragma unroll
                        for (int o = 16; o > 0; o >>= 1) {
                            float ov = __shfl_xor(lv, o, 32);
                            int   oi = __shfl_xor(li, o, 32);
                            if (ov < lv) { lv = ov; li = oi; }
                        }
                        if (lane == 0) rowTop[li] = cand;
                        // recompute threshold (min of updated set)
                        float l2 = cand;
                        for (int j = lane; j < TOPK_K; j += 32) l2 = fminf(l2, rowTop[j]);
#pragma unroll
                        for (int o = 16; o > 0; o >>= 1) l2 = fminf(l2, __shfl_xor(l2, o, 32));
                        vm = l2;
                    }
                }
            }
            vmin[rr] = vm;
        }
        __syncthreads();
    }
    // mean of top-100: wave w writes rows {w, w+16, w+32, w+48}
#pragma unroll
    for (int rr = 0; rr < 4; ++rr) {
        int row = wave + rr * 16;
        float s = 0.0f;
        for (int j = lane; j < TOPK_K; j += 32) s += top[row * TOPK_K + j];
#pragma unroll
        for (int o = 16; o > 0; o >>= 1) s += __shfl_xor(s, o, 32);
        if (lane == 0) A1[row0 + row] = s * (1.0f / TOPK_K);
    }
}

// ---------------- Kernel 3: stable ascending rank -> sel[rank]=i for rank < M_SEL ----------------
__global__ __launch_bounds__(256) void k_rank(const float* __restrict__ A1, int* __restrict__ sel) {
    __shared__ float a1s[2048];
    int i = blockIdx.x * 256 + threadIdx.x;
    float myv = A1[i];
    int rank = 0;
    for (int c0 = 0; c0 < N_TOK; c0 += 2048) {
        for (int j = threadIdx.x; j < 2048; j += 256) a1s[j] = A1[c0 + j];
        __syncthreads();
        for (int j = 0; j < 2048; ++j) {
            float o = a1s[j];
            int jg = c0 + j;
            rank += (o < myv) || (o == myv && jg < i);
        }
        __syncthreads();
    }
    if (rank < M_SEL) sel[rank] = i;
}

// ---------------- Kernel 4: feat = relu( x0[sel] @ W1 + b1 ), WMMA with row gather ----------------
__global__ __launch_bounds__(256) void k_feat(const float* __restrict__ x0,
                                              const int* __restrict__ sel,
                                              const _Float16* __restrict__ W1T,
                                              const float* __restrict__ b1,
                                              float* __restrict__ feat) {
    __shared__ int srow[16];
    int tid = threadIdx.x, wave = tid >> 5, lane = tid & 31;
    int m0 = blockIdx.x * 16;
    if (tid < 16) {
        int mg = m0 + tid;
        srow[tid] = sel[mg < M_SEL ? mg : (M_SEL - 1)];
    }
    __syncthreads();
    int m = lane & 15, n = lane & 15;
    const float* xrow = x0 + (size_t)srow[m] * DIM;

    v8f acc[4];
#pragma unroll
    for (int t = 0; t < 4; ++t) acc[t] = (v8f){};
    for (int k0 = 0; k0 < DIM; k0 += 32) {
        v16h a = load_a_f32(xrow, k0, lane);
#pragma unroll
        for (int t = 0; t < 4; ++t) {
            int n0 = (wave * 4 + t) * 16;
            v16h b = load_b_f16(W1T + (size_t)(n0 + n) * DIM, k0, lane);
            acc[t] = wmma_f16(a, b, acc[t]);
        }
    }
#pragma unroll
    for (int t = 0; t < 4; ++t) {
        int n0 = (wave * 4 + t) * 16;
        float bb = b1[n0 + n];
#pragma unroll
        for (int r = 0; r < 8; ++r) {
            int mm = r + (lane < 16 ? 0 : 8);
            int mg = m0 + mm;
            if (mg < M_SEL) {
                float v = acc[t][r] + bb;
                feat[(size_t)mg * HD + n0 + n] = fmaxf(v, 0.0f);
            }
        }
    }
}

// ---------------- Kernel 5: logits[r] = (tanh(f@Wv+bv) * sigmoid(f@Wu+bu)) @ Wa + ba ----------------
__global__ __launch_bounds__(256) void k_attlogit(const float* __restrict__ feat,
                                                  const float* __restrict__ Wv, const float* __restrict__ bvv,
                                                  const float* __restrict__ Wu, const float* __restrict__ bu,
                                                  const float* __restrict__ Wa, const float* __restrict__ ba,
                                                  float* __restrict__ logits) {
    __shared__ float fs[HD];
    __shared__ float red[256];
    int r = blockIdx.x, t = threadIdx.x;
    for (int j = t; j < HD; j += 256) fs[j] = feat[(size_t)r * HD + j];
    __syncthreads();
    float s1 = bvv[t], s2 = bu[t];
    for (int k = 0; k < HD; ++k) {
        float f = fs[k];
        s1 += f * Wv[k * HD2 + t];
        s2 += f * Wu[k * HD2 + t];
    }
    float a = tanhf(s1) * (1.0f / (1.0f + expf(-s2))) * Wa[t];
    red[t] = a;
    __syncthreads();
    for (int o = 128; o > 0; o >>= 1) {
        if (t < o) red[t] += red[t + o];
        __syncthreads();
    }
    if (t == 0) logits[r] = red[0] + ba[0];
}

// ---------------- Kernel 6: softmax over M_SEL -> att (written straight to d_out+1) ----------------
__global__ __launch_bounds__(256) void k_softmax(const float* __restrict__ logits,
                                                 float* __restrict__ att_out) {
    __shared__ float red[256];
    int t = threadIdx.x;
    float mx = -3.0e38f;
    for (int i = t; i < M_SEL; i += 256) mx = fmaxf(mx, logits[i]);
    red[t] = mx; __syncthreads();
    for (int o = 128; o > 0; o >>= 1) { if (t < o) red[t] = fmaxf(red[t], red[t + o]); __syncthreads(); }
    mx = red[0];
    __syncthreads();
    float s = 0.0f;
    for (int i = t; i < M_SEL; i += 256) s += expf(logits[i] - mx);
    red[t] = s; __syncthreads();
    for (int o = 128; o > 0; o >>= 1) { if (t < o) red[t] += red[t + o]; __syncthreads(); }
    float inv = 1.0f / red[0];
    for (int i = t; i < M_SEL; i += 256) att_out[i] = expf(logits[i] - mx) * inv;
}

// ---------------- Kernel 7: Mv = att @ feat ; Y = Mv @ Wc + bc ----------------
__global__ __launch_bounds__(512) void k_mv(const float* __restrict__ att,
                                            const float* __restrict__ feat,
                                            const float* __restrict__ Wc,
                                            const float* __restrict__ bc,
                                            float* __restrict__ y) {
    __shared__ float ac[512];
    __shared__ float red[512];
    int t = threadIdx.x;
    float mv = 0.0f;
    for (int r0 = 0; r0 < M_SEL; r0 += 512) {
        int nr = (M_SEL - r0) < 512 ? (M_SEL - r0) : 512;
        if (t < nr) ac[t] = att[r0 + t];
        __syncthreads();
        for (int r = 0; r < nr; ++r) mv += ac[r] * feat[(size_t)(r0 + r) * HD + t];
        __syncthreads();
    }
    red[t] = mv * Wc[t];
    __syncthreads();
    for (int o = 256; o > 0; o >>= 1) { if (t < o) red[t] += red[t + o]; __syncthreads(); }
    if (t == 0) y[0] = red[0] + bc[0];
}

extern "C" void kernel_launch(void* const* d_in, const int* in_sizes, int n_in,
                              void* d_out, int out_size, void* d_ws, size_t ws_size,
                              hipStream_t stream) {
    const float* x0 = (const float*)d_in[0];
    const float* x1 = (const float*)d_in[1];
    const float* Wq = (const float*)d_in[2];
    const float* bq = (const float*)d_in[3];
    const float* Wk = (const float*)d_in[4];
    const float* bk = (const float*)d_in[5];
    const float* W1 = (const float*)d_in[6];
    const float* b1 = (const float*)d_in[7];
    const float* Wv = (const float*)d_in[8];
    const float* bvv = (const float*)d_in[9];
    const float* Wu = (const float*)d_in[10];
    const float* bu = (const float*)d_in[11];
    const float* Wa = (const float*)d_in[12];
    const float* ba = (const float*)d_in[13];
    const float* Wc = (const float*)d_in[14];
    const float* bc = (const float*)d_in[15];

    char* ws = (char*)d_ws;
    _Float16* qh  = (_Float16*)(ws + 0);                 // 16 MiB
    _Float16* kh  = (_Float16*)(ws + (16u << 20));       // 16 MiB
    _Float16* WqT = (_Float16*)(ws + (32u << 20));       // 1 MiB
    _Float16* WkT = (_Float16*)(ws + (33u << 20));       // 1 MiB
    _Float16* W1T = (_Float16*)(ws + (34u << 20));       // 1 MiB
    float*    A1  = (float*)   (ws + (35u << 20));       // 64 KiB
    int*      sel = (int*)     (ws + (36u << 20));       // ~20 KiB
    float*    feat= (float*)   (ws + (37u << 20));       // ~10 MiB
    float*    logits = (float*)(ws + (48u << 20));       // ~20 KiB
    float*    out = (float*)d_out;                       // [0]=Y, [1..4915]=att

    k_wt<<<2048, 256, 0, stream>>>(Wq, WqT);
    k_wt<<<2048, 256, 0, stream>>>(Wk, WkT);
    k_wt<<<2048, 256, 0, stream>>>(W1, W1T);

    k_proj<<<N_TOK / 16, 256, 0, stream>>>(x0, WqT, bq, qh);
    k_proj<<<N_TOK / 16, 256, 0, stream>>>(x1, WkT, bk, kh);

    k_sim_topk<<<N_TOK / SIM_ROWS, 512, SIM_LDS_BYTES, stream>>>(qh, kh, A1);

    k_rank<<<N_TOK / 256, 256, 0, stream>>>(A1, sel);

    k_feat<<<(M_SEL + 15) / 16, 256, 0, stream>>>(x0, sel, W1T, b1, feat);

    k_attlogit<<<M_SEL, 256, 0, stream>>>(feat, Wv, bvv, Wu, bu, Wa, ba, logits);

    k_softmax<<<1, 256, 0, stream>>>(logits, out + 1);

    k_mv<<<1, 512, 0, stream>>>(out + 1, feat, Wc, bc, out);
}
